// LSTM_28501402976270
// MI455X (gfx1250) — compile-verified, matrix operand order checked by reading, not run on hardware
//
#include <hip/hip_runtime.h>
#include <hip/hip_bf16.h>
#include <stdint.h>

// Problem sizes (fixed by the reference)
#define V_    32000
#define H_    512
#define B_    8
#define S_    512
#define FOURH 2048          // 4*H
#define MROWS 4096          // B*S

typedef __attribute__((ext_vector_type(16))) __bf16 v16bf;
typedef __attribute__((ext_vector_type(8)))  float  v8f;
typedef int vi4 __attribute__((vector_size(16)));   // matches builtin param pointee

union FragU {
  v16bf v;
  uint4 q[2];
};

// Async global->LDS path (CDNA5 GLOBAL_LOAD_ASYNC_TO_LDS_B128, ASYNCcnt),
// guarded so we fall back to the synchronous path if the toolchain lacks it.
#if defined(__has_builtin)
#if __has_builtin(__builtin_amdgcn_global_load_async_to_lds_b128) && \
    __has_builtin(__builtin_amdgcn_s_wait_asynccnt)
#define USE_ASYNC_LDS 1
#endif
#endif
#ifndef USE_ASYNC_LDS
#define USE_ASYNC_LDS 0
#endif

#if USE_ASYNC_LDS
typedef __attribute__((address_space(1))) vi4 g_vi4;
typedef __attribute__((address_space(3))) vi4 l_vi4;
#endif

__device__ __forceinline__ unsigned short f2bf(float f) {
  union { float f; unsigned u; } a; a.f = f;
  unsigned u = a.u;
  unsigned r = (u + 0x7FFFu + ((u >> 16) & 1u)) >> 16;   // round-to-nearest-even
  return (unsigned short)r;
}

// ---------------------------------------------------------------------------
// Embedding gather -> bf16 X [MROWS, H] (row-major; A operand of the Gx GEMM)
// ---------------------------------------------------------------------------
__global__ void k_embed(const int* __restrict__ tok, const float* __restrict__ emb,
                        unsigned short* __restrict__ Xbf) {
  int idx = blockIdx.x * blockDim.x + threadIdx.x;   // MROWS*H elements
  int row = idx >> 9;                                // / H (=512)
  int h   = idx & (H_ - 1);
  int t   = tok[row];
  Xbf[idx] = f2bf(emb[(size_t)t * H_ + h]);
}

// ---------------------------------------------------------------------------
// Swizzle a [K=512, N] f32 weight matrix into per-lane WMMA B-fragment layout.
// Fragment (ntile, ktile): lane L holds 16 bf16 for column n = ntile*16 + (L&15),
// element e maps to K = ktile*32 + (e&7) + 8*(L>>4) + 16*(e>>3)
// (ISA 7.12.2 16-bit operand layout). Stored contiguously so consumers load
// each lane's fragment as two aligned global b128s (L2-resident weights).
// ---------------------------------------------------------------------------
__global__ void k_swizB(const float* __restrict__ src, int rowStride,
                        unsigned short* __restrict__ dst) {
  int idx  = blockIdx.x * blockDim.x + threadIdx.x;  // N*512 elements
  int e    = idx & 15;
  int lane = (idx >> 4) & 31;
  int kt   = (idx >> 9) & 15;                        // K/32 = 16 ktiles
  int nt   = idx >> 13;
  int laneHi = lane >> 4;
  int k = kt * 32 + (e & 7) + 8 * laneHi + 16 * (e >> 3);
  int n = nt * 16 + (lane & 15);
  dst[idx] = f2bf(src[(size_t)k * rowStride + n]);
}

// ---------------------------------------------------------------------------
// Tiled bf16 GEMM, f32 accumulation: C[M,N] = A[M,K] @ B[K,N] + bias[N].
// A [M,K] bf16 row-major staged through LDS (async when available);
// B pre-swizzled fragments loaded directly from global (L2-resident).
// Block tile 128x128, 256 threads = 8 waves, wave tile 64x32 (4x2 of 16x16).
// ---------------------------------------------------------------------------
#define BM  128
#define BN  128
#define KT  32
#define LDP 40      // padded LDS row stride (bf16 elems): 80B, 16B aligned, bank-skewed

__global__ __launch_bounds__(256)
void k_gemm(const unsigned short* __restrict__ A,      // [M,K] bf16 row-major
            const unsigned short* __restrict__ Bfrag,  // fragment-swizzled [K,N]
            const float* __restrict__ bias,            // [N]
            float* __restrict__ C,                     // [M,N] f32
            int M, int N, int K) {
  __shared__ unsigned short At[BM * LDP];

  const int tid    = threadIdx.x;
  const int lane   = tid & 31;
  const int wave   = tid >> 5;       // 0..7
  const int wm     = wave & 1;       // 2 waves along M
  const int wn     = wave >> 1;      // 4 waves along N
  const int laneHi = lane >> 4;
  const int lane15 = lane & 15;

  const int row0   = blockIdx.x * BM;
  const int n0     = blockIdx.y * BN;
  const int ktiles = K >> 5;
  const int nt0    = (n0 >> 4) + wn * 2;   // this wave's first global ntile

  v8f acc[4][2] = {};

  for (int k0 = 0; k0 < K; k0 += KT) {
    const int kt = k0 >> 5;
    // --- stage A tile: 128 rows x 32 bf16 = 512 b128 chunks, 2 per thread ---
#pragma unroll
    for (int c = 0; c < 2; ++c) {
      int ch = tid + c * 256;
      int r  = ch >> 2, q = ch & 3;
      const unsigned short* gp = A + (size_t)(row0 + r) * K + k0 + q * 8;
      unsigned short*       lp = &At[r * LDP + q * 8];
#if USE_ASYNC_LDS
      __builtin_amdgcn_global_load_async_to_lds_b128(
          (g_vi4*)gp, (l_vi4*)lp, 0, 0);
#else
      *(uint4*)lp = *(const uint4*)gp;
#endif
    }
#if USE_ASYNC_LDS
    __builtin_amdgcn_s_wait_asynccnt(0);
#endif
    __syncthreads();

    // --- B fragments: two aligned b128 global loads each (hits L2) ---
    FragU bfr[2];
#pragma unroll
    for (int j = 0; j < 2; ++j) {
      const unsigned short* bp =
          Bfrag + (((size_t)(nt0 + j) * ktiles + kt) * 32 + lane) * 16;
      bfr[j].q[0] = ((const uint4*)bp)[0];
      bfr[j].q[1] = ((const uint4*)bp)[1];
      if (k0 + KT < K) __builtin_prefetch(bp + 512, 0, 1);   // next ktile frag
    }
    // --- A fragments from LDS: two aligned b128 ds loads each ---
    FragU af[4];
#pragma unroll
    for (int i = 0; i < 4; ++i) {
      int r = wm * 64 + i * 16 + lane15;
      const unsigned short* p = &At[r * LDP + 8 * laneHi];
      af[i].q[0] = *(const uint4*)(p);
      af[i].q[1] = *(const uint4*)(p + 16);
    }
#pragma unroll
    for (int i = 0; i < 4; ++i)
#pragma unroll
      for (int j = 0; j < 2; ++j)
        acc[i][j] = __builtin_amdgcn_wmma_f32_16x16x32_bf16(
            false, af[i].v, false, bfr[j].v, (short)0, acc[i][j], false, false);
    __syncthreads();
  }

  // --- epilogue: C/D layout — VGPR v, lane: row = v + 8*laneHi, col = lane&15 ---
#pragma unroll
  for (int i = 0; i < 4; ++i) {
    int rbase = row0 + wm * 64 + i * 16 + 8 * laneHi;
#pragma unroll
    for (int j = 0; j < 2; ++j) {
      int   col = n0 + wn * 32 + j * 16 + lane15;
      float bv  = bias[col];
#pragma unroll
      for (int v = 0; v < 8; ++v)
        C[(size_t)(rbase + v) * N + col] = acc[i][j][v] + bv;
    }
  }
}

// ---------------------------------------------------------------------------
// Sequential LSTM recurrence. One persistent workgroup (32 wave32s on a WGP).
// Per step: gates[8,2048] = h[8,512] @ Wh + Gx[:,s,:]  (Gx already has x@Wx+b),
// computed with WMMA (M=16 tile, rows 8..15 zero); then elementwise c/h update.
// Wh comes pre-swizzled so each B fragment is two b128 loads (L2-resident).
// ---------------------------------------------------------------------------
#define HPAD 520   // padded hbf row stride (bf16): skews banks, keeps 16B align

__global__ __launch_bounds__(1024)
void k_lstm(const float* __restrict__ Gx,              // [MROWS, 4H] f32
            const unsigned short* __restrict__ WhT,    // fragment-swizzled Wh
            unsigned short* __restrict__ HSbf) {       // [MROWS, H] bf16 out
  __shared__ unsigned short hbf[16 * HPAD];            // bf16 h, rows 8..15 = 0
  __shared__ float cst[B_ * H_];                       // f32 cell state
  __shared__ float gates[B_ * FOURH];                  // f32 gate values

  const int tid    = threadIdx.x;
  const int lane   = tid & 31;
  const int wave   = tid >> 5;     // 0..31
  const int laneHi = lane >> 4;
  const int lane15 = lane & 15;

  for (int i = tid; i < 16 * HPAD; i += 1024) hbf[i] = 0;
  for (int i = tid; i < B_ * H_;   i += 1024) cst[i] = 0.f;
  __syncthreads();

  for (int s = 0; s < S_; ++s) {
    // ---- gate GEMM: each wave owns 4 N-tiles (64 cols), K=512 in 16 chunks ----
    v8f acc[4] = {};
    for (int kc = 0; kc < 16; ++kc) {
      FragU af;
      const unsigned short* p = &hbf[lane15 * HPAD + kc * 32 + 8 * laneHi];
      af.q[0] = *(const uint4*)(p);
      af.q[1] = *(const uint4*)(p + 16);
#pragma unroll
      for (int t = 0; t < 4; ++t) {
        int nt = wave * 4 + t;
        FragU bfr;
        const uint4* g = (const uint4*)(WhT + ((size_t)(nt * 16 + kc) * 32 + lane) * 16);
        bfr.q[0] = g[0];
        bfr.q[1] = g[1];
        acc[t] = __builtin_amdgcn_wmma_f32_16x16x32_bf16(
            false, af.v, false, bfr.v, (short)0, acc[t], false, false);
      }
    }
    // ---- add Gx and spill gates to LDS (rows 0..7 = batch; 8..15 discarded) ----
#pragma unroll
    for (int t = 0; t < 4; ++t) {
      int col = (wave * 4 + t) * 16 + lane15;
      if (laneHi == 0) {
#pragma unroll
        for (int v = 0; v < 8; ++v)
          gates[v * FOURH + col] =
              acc[t][v] + Gx[((size_t)v * S_ + s) * FOURH + col];
      }
    }
    __syncthreads();

    // ---- elementwise LSTM cell: 4096 (b,j) units over 1024 threads ----
    {
      int u0 = tid * 4;
      int b  = u0 >> 9;           // / H
      int j0 = u0 & (H_ - 1);
#pragma unroll
      for (int q = 0; q < 4; ++q) {
        int   j = j0 + q;
        float f = gates[b * FOURH +           j];
        float i = gates[b * FOURH +     H_ +  j];
        float o = gates[b * FOURH + 2 * H_ +  j];
        float g = gates[b * FOURH + 3 * H_ +  j];
        f = 1.f / (1.f + __expf(-f));
        i = 1.f / (1.f + __expf(-i));
        o = 1.f / (1.f + __expf(-o));
        g = tanhf(g);
        float c = f * cst[b * H_ + j] + i * g;
        cst[b * H_ + j] = c;
        float h = o * tanhf(c);
        unsigned short hb = f2bf(h);
        hbf[b * HPAD + j] = hb;
        HSbf[((size_t)b * S_ + s) * H_ + j] = hb;
      }
    }
    __syncthreads();
  }
}

// ---------------------------------------------------------------------------
// Host-side launcher
// ---------------------------------------------------------------------------
extern "C" void kernel_launch(void* const* d_in, const int* in_sizes, int n_in,
                              void* d_out, int out_size, void* d_ws, size_t ws_size,
                              hipStream_t stream) {
  const int*   tok   = (const int*)  d_in[0];   // [B,S]
  const float* emb   = (const float*)d_in[1];   // [V,H]
  const float* W     = (const float*)d_in[2];   // [2H,4H]
  const float* bgate = (const float*)d_in[3];   // [4H]
  const float* Wout  = (const float*)d_in[4];   // [H,V]
  const float* bout  = (const float*)d_in[5];   // [V]
  float*       out   = (float*)d_out;           // [B,S,V]

  char* ws = (char*)d_ws;
  unsigned short* Xbf    = (unsigned short*)ws; ws += (size_t)MROWS * H_ * 2;      //  4.0 MB
  unsigned short* Wxbf   = (unsigned short*)ws; ws += (size_t)H_ * FOURH * 2;      //  2.0 MB
  unsigned short* WhT    = (unsigned short*)ws; ws += (size_t)H_ * FOURH * 2;      //  2.0 MB
  unsigned short* Woutbf = (unsigned short*)ws; ws += (size_t)H_ * V_ * 2;         // 32.8 MB
  unsigned short* HSbf   = (unsigned short*)ws; ws += (size_t)MROWS * H_ * 2;      //  4.0 MB
  float*          Gx     = (float*)ws;                                             // 33.6 MB

  // 1) embedding gather -> bf16
  k_embed<<<(MROWS * H_) / 256, 256, 0, stream>>>(tok, emb, Xbf);
  // 2) weight swizzles into WMMA B-fragment layout (one-time, L2-resident after)
  k_swizB<<<(H_ * FOURH) / 256, 256, 0, stream>>>(W,               FOURH, Wxbf);
  k_swizB<<<(H_ * FOURH) / 256, 256, 0, stream>>>(W + H_ * FOURH,  FOURH, WhT);
  k_swizB<<<(H_ * V_)    / 256, 256, 0, stream>>>(Wout,            V_,    Woutbf);
  // 3) hoisted input half of the gates: Gx = X @ Wx + b   (M=4096,N=2048,K=512)
  {
    dim3 g(MROWS / BM, FOURH / BN);
    k_gemm<<<g, 256, 0, stream>>>(Xbf, Wxbf, bgate, Gx, MROWS, FOURH, H_);
  }
  // 4) sequential recurrence (one persistent WGP)
  k_lstm<<<1, 1024, 0, stream>>>(Gx, WhT, HSbf);
  // 5) output projection: logits = hs @ Wout + bout   (M=4096,N=32000,K=512)
  {
    dim3 g(MROWS / BM, V_ / BN);
    k_gemm<<<g, 256, 0, stream>>>(HSbf, Woutbf, bout, out, MROWS, V_, H_);
  }
}
